// SpERT_70909910057328
// MI455X (gfx1250) — matrix operand order repaired and checked by reading, not compiled, not run on hardware
//
#include <hip/hip_runtime.h>

#define NB 2
#define NS 128
#define NL 512
#define NH 768
#define NE 25
#define NT_ENT 5
#define NT_REL 6
#define K1 1561            // H + H + E
#define KP1 1568           // padded to 32
#define K2 2354            // H + 2H + 2E
#define KP2 2368           // padded to 32

typedef __attribute__((ext_vector_type(16))) __bf16 v16bf;
typedef __attribute__((ext_vector_type(8)))  __bf16 v8bf;
typedef __attribute__((ext_vector_type(8)))  float  v8f;

__device__ inline __bf16 f2bf(float x) {
    union { float f; unsigned u; } in; in.f = x;
    unsigned r = (in.u + 0x7FFFu + ((in.u >> 16) & 1u)) >> 16;
    union { unsigned short s; __bf16 b; } out; out.s = (unsigned short)r;
    return out.b;
}

// ---------------------------------------------------------------------------
// Masked max-pool over L. Spans are contiguous (reference _span_masks), so we
// reduce (first-set-index, popcount) once, then loop only over [start,
// start+len) with unconditional coalesced loads. Accumulator starts at 0,
// matching the reference's max over (h * mask) which always includes zeros.
// blockIdx.y: 0 = entity_mask (also emits span length), 1 = rel context mask.
// One block per (b, span).
// ---------------------------------------------------------------------------
__global__ __launch_bounds__(256) void masked_max_kernel(
    const float* __restrict__ h, const int* __restrict__ emask,
    const int* __restrict__ rmask, float* __restrict__ ent,
    float* __restrict__ rctx, int* __restrict__ elen) {
    __shared__ int psum[256];
    __shared__ int pmin[256];
    const int bs  = blockIdx.x;            // 0..B*S-1
    const int b   = bs / NS;
    const int tid = threadIdx.x;
    const int* mask = (blockIdx.y == 0 ? emask : rmask) + bs * NL;
    const int m0 = mask[tid];
    const int m1 = mask[tid + 256];
    psum[tid] = m0 + m1;
    pmin[tid] = m0 ? tid : (m1 ? tid + 256 : NL);
    __syncthreads();
    for (int s = 128; s > 0; s >>= 1) {
        if (tid < s) {
            psum[tid] += psum[tid + s];
            pmin[tid] = min(pmin[tid], pmin[tid + s]);
        }
        __syncthreads();
    }
    const int len   = psum[0];
    const int start = pmin[0];

    const float* hb = h + (size_t)b * NL * NH;
    float a0 = 0.f, a1 = 0.f, a2 = 0.f;
    for (int l = start; l < start + len; ++l) {
        const float* row = hb + (size_t)l * NH;
        a0 = fmaxf(a0, row[tid]);
        a1 = fmaxf(a1, row[tid + 256]);
        a2 = fmaxf(a2, row[tid + 512]);
    }
    float* out = (blockIdx.y == 0 ? ent : rctx) + (size_t)bs * NH;
    out[tid] = a0; out[tid + 256] = a1; out[tid + 512] = a2;

    if (tid == 0 && blockIdx.y == 0) elen[bs] = len;
}

// ---------------------------------------------------------------------------
// Build bf16 entity_repr rows: [entity(H) | ctx(H) | size_emb(E) | pad]
// ---------------------------------------------------------------------------
__global__ __launch_bounds__(256) void pack_ent_kernel(
    const float* __restrict__ ent, const float* __restrict__ h,
    const float* __restrict__ size_emb, const int* __restrict__ elen,
    __bf16* __restrict__ A) {
    const int row = blockIdx.x;            // 0..255 = b*S+s
    const int b   = row / NS;
    const int len = elen[row];
    for (int k = threadIdx.x; k < KP1; k += 256) {
        float v;
        if      (k < NH)          v = ent[(size_t)row * NH + k];
        else if (k < 2 * NH)      v = h[(size_t)b * NL * NH + (k - NH)];   // h[b,0,:]
        else if (k < 2 * NH + NE) v = size_emb[len * NE + (k - 2 * NH)];
        else                      v = 0.f;
        A[(size_t)row * KP1 + k] = f2bf(v);
    }
}

// ---------------------------------------------------------------------------
// Build bf16 rel_repr rows: [rel_ctx(H) | ent[e1](H) | ent[e2](H) |
//                            size[e1](E) | size[e2](E) | pad]
// ---------------------------------------------------------------------------
__global__ __launch_bounds__(256) void pack_rel_kernel(
    const float* __restrict__ rctx, const float* __restrict__ ent,
    const float* __restrict__ size_emb, const int* __restrict__ elen,
    const int* __restrict__ relations, __bf16* __restrict__ A) {
    const int row = blockIdx.x;            // 0..255 = b*R+r
    const int b   = row / NS;
    const int e1  = b * NS + relations[row * 2 + 0];
    const int e2  = b * NS + relations[row * 2 + 1];
    const int l1  = elen[e1], l2 = elen[e2];
    for (int k = threadIdx.x; k < KP2; k += 256) {
        float v;
        if      (k < NH)               v = rctx[(size_t)row * NH + k];
        else if (k < 2 * NH)           v = ent[(size_t)e1 * NH + (k - NH)];
        else if (k < 3 * NH)           v = ent[(size_t)e2 * NH + (k - 2 * NH)];
        else if (k < 3 * NH + NE)      v = size_emb[l1 * NE + (k - 3 * NH)];
        else if (k < 3 * NH + 2 * NE)  v = size_emb[l2 * NE + (k - 3 * NH - NE)];
        else                           v = 0.f;
        A[(size_t)row * KP2 + k] = f2bf(v);
    }
}

// ---------------------------------------------------------------------------
// Pack weights (NT x K row-major, fp32) into the WMMA B-matrix lane layout,
// pre-swizzled so each lane reads one contiguous v16bf per 32-K tile:
//   Bp[((t*32 + lane)*16 + j)] = W[n][ t*32 + (lane>>4)*8 + (j&7) + (j>>3)*16 ],  n = lane&15
// ---------------------------------------------------------------------------
__global__ __launch_bounds__(256) void pack_w_kernel(
    const float* __restrict__ w, int nt, int K, int Kp, __bf16* __restrict__ Bp) {
    const int idx = blockIdx.x * 256 + threadIdx.x;
    if (idx >= Kp * 16) return;
    const int j        = idx & 15;
    const int lanetile = idx >> 4;
    const int lane     = lanetile & 31;
    const int t        = lanetile >> 5;
    const int k        = t * 32 + ((lane >> 4) * 8) + (j & 7) + ((j >> 3) * 16);
    const int n        = lane & 15;
    const float v = (n < nt && k < K) ? w[n * K + k] : 0.f;
    Bp[idx] = f2bf(v);
}

// ---------------------------------------------------------------------------
// GEMM: out(256 x nt) = A(256 x Kp, bf16) * Bp(WMMA-swizzled) + bias.
// One wave per 16-row tile. A per ISA 16-bit 16x32 layout:
//   lane m = lane&15; hi = lane>>4; elems 0..7 = A[m][kb + hi*8 + 0..7],
//   elems 8..15 = A[m][kb + hi*8 + 16 + 0..7]  -> two contiguous 16B loads.
// D layout: n = lane&15, row = hi*8 + r for acc element r.
// ---------------------------------------------------------------------------
__global__ __launch_bounds__(32) void wmma_gemm_kernel(
    const __bf16* __restrict__ A, const __bf16* __restrict__ Bp,
    const float* __restrict__ bias, float* __restrict__ out,
    int Kp, int nt) {
    const int lane = threadIdx.x;
    const int tile = blockIdx.x;
    const int m    = lane & 15;
    const int hi   = lane >> 4;
    const __bf16* arow = A + (size_t)(tile * 16 + m) * Kp + hi * 8;

    v8f acc = {};
    for (int kb = 0; kb < Kp; kb += 32) {
        const v8bf alo = *(const v8bf*)(arow + kb);
        const v8bf ahi = *(const v8bf*)(arow + kb + 16);
        v16bf a;
#pragma unroll
        for (int j = 0; j < 8; ++j) { a[j] = alo[j]; a[8 + j] = ahi[j]; }
        const v16bf bv = *(const v16bf*)(Bp + (size_t)(kb + lane) * 16);
        acc = __builtin_amdgcn_wmma_f32_16x16x32_bf16(
            false, a, false, bv, (short)0, acc, false, false);
    }

    const int n = lane & 15;
    if (n < nt) {
        const float bb = bias[n];
#pragma unroll
        for (int r = 0; r < 8; ++r) {
            const int mg = tile * 16 + hi * 8 + r;
            out[mg * nt + n] = acc[r] + bb;
        }
    }
}

// ---------------------------------------------------------------------------
extern "C" void kernel_launch(void* const* d_in, const int* in_sizes, int n_in,
                              void* d_out, int out_size, void* d_ws, size_t ws_size,
                              hipStream_t stream) {
    const float* h         = (const float*)d_in[0];
    const int*   emask     = (const int*)  d_in[1];
    const int*   relations = (const int*)  d_in[2];
    const int*   rmask     = (const int*)  d_in[3];
    const float* size_emb  = (const float*)d_in[4];
    const float* span_w    = (const float*)d_in[5];
    const float* span_b    = (const float*)d_in[6];
    const float* rel_w     = (const float*)d_in[7];
    const float* rel_b     = (const float*)d_in[8];

    char* ws = (char*)d_ws;
    const size_t OFF_ENT  = 0;                                   // 256*768 f32
    const size_t OFF_RCTX = OFF_ENT  + (size_t)NB * NS * NH * 4;
    const size_t OFF_LEN  = OFF_RCTX + (size_t)NB * NS * NH * 4; // 256 int
    const size_t OFF_AE   = OFF_LEN  + 1024;                     // 256*KP1 bf16
    const size_t OFF_AR   = OFF_AE   + (size_t)NB * NS * KP1 * 2;
    const size_t OFF_BE   = OFF_AR   + (size_t)NB * NS * KP2 * 2;
    const size_t OFF_BR   = OFF_BE   + (size_t)KP1 * 16 * 2;

    float*  ent  = (float*)(ws + OFF_ENT);
    float*  rctx = (float*)(ws + OFF_RCTX);
    int*    elen = (int*)  (ws + OFF_LEN);
    __bf16* Aent = (__bf16*)(ws + OFF_AE);
    __bf16* Arel = (__bf16*)(ws + OFF_AR);
    __bf16* Bent = (__bf16*)(ws + OFF_BE);
    __bf16* Brel = (__bf16*)(ws + OFF_BR);

    masked_max_kernel<<<dim3(NB * NS, 2), 256, 0, stream>>>(
        h, emask, rmask, ent, rctx, elen);
    pack_ent_kernel<<<NB * NS, 256, 0, stream>>>(ent, h, size_emb, elen, Aent);
    pack_rel_kernel<<<NB * NS, 256, 0, stream>>>(rctx, ent, size_emb, elen,
                                                 relations, Arel);
    pack_w_kernel<<<(KP1 * 16 + 255) / 256, 256, 0, stream>>>(
        span_w, NT_ENT, K1, KP1, Bent);
    pack_w_kernel<<<(KP2 * 16 + 255) / 256, 256, 0, stream>>>(
        rel_w, NT_REL, K2, KP2, Brel);

    float* out_ent = (float*)d_out;                 // (B,S,5) flat
    float* out_rel = out_ent + NB * NS * NT_ENT;    // (B,R,6) flat
    wmma_gemm_kernel<<<(NB * NS) / 16, 32, 0, stream>>>(
        Aent, Bent, span_b, out_ent, KP1, NT_ENT);
    wmma_gemm_kernel<<<(NB * NS) / 16, 32, 0, stream>>>(
        Arel, Brel, rel_b, out_rel, KP2, NT_REL);
}
